// Attention_22789096473341
// MI455X (gfx1250) — matrix operand order repaired and checked by reading, not compiled, not run on hardware
//
#include <hip/hip_runtime.h>

#define SEQ    2048
#define DIM    4096
#define NQH    32
#define NKVH   8
#define HD     128
#define KVDIM  (NKVH * HD)   // 1024
#define WINDOW 1024

#if __has_builtin(__builtin_amdgcn_global_load_async_to_lds_b128) && \
    __has_builtin(__builtin_amdgcn_s_wait_asynccnt)
#define HAVE_ASYNC_LDS 1
#endif

typedef __attribute__((ext_vector_type(8)))  __bf16 v8bf;
typedef __attribute__((ext_vector_type(16))) __bf16 v16bf;
typedef __attribute__((ext_vector_type(8)))  float  v8f;
typedef __attribute__((__vector_size__(16))) int    v4i;

#ifdef HAVE_ASYNC_LDS
typedef __attribute__((address_space(1))) v4i as1_v4i;   // global
typedef __attribute__((address_space(3))) v4i as3_v4i;   // LDS
#endif

static __device__ __forceinline__ v16bf cat16(v8bf lo, v8bf hi) {
  return __builtin_shufflevector(lo, hi, 0,1,2,3,4,5,6,7,8,9,10,11,12,13,14,15);
}
static __device__ __forceinline__ v8f wmma_bf16(v16bf a, v16bf b, v8f c) {
  // D = A(16x32) * B(32x16) + C, f32 accumulate
  return __builtin_amdgcn_wmma_f32_16x16x32_bf16(false, a, false, b, (short)0, c, false, false);
}

// ---------------- elementwise / layout kernels ----------------

__global__ void cvt_kernel(const float* __restrict__ in, __bf16* __restrict__ out, long n) {
  long i = (long)blockIdx.x * blockDim.x + threadIdx.x;
  if (i < n) out[i] = (__bf16)in[i];
}

// in: fp32 [k_dim][n_dim] row-major -> out: bf16 [n_dim][k_dim]  (weight transpose)
__global__ void cvt_t_kernel(const float* __restrict__ in, __bf16* __restrict__ out,
                             int k_dim, int n_dim) {
  long i = (long)blockIdx.x * blockDim.x + threadIdx.x;
  long total = (long)k_dim * n_dim;
  if (i >= total) return;
  int k = (int)(i / n_dim);
  int n = (int)(i % n_dim);
  out[(long)n * k_dim + k] = (__bf16)in[i];
}

// buf: bf16 [SEQ][nheads][HD], rotate pairs along HD with freqs [SEQ][HD/2]
__global__ void rope_kernel(__bf16* __restrict__ buf, const float* __restrict__ fc,
                            const float* __restrict__ fs, int nheads) {
  long i = (long)blockIdx.x * blockDim.x + threadIdx.x;
  long total = (long)SEQ * nheads * (HD / 2);
  if (i >= total) return;
  int p = (int)(i % (HD / 2));
  long t = i / (HD / 2);
  int h = (int)(t % nheads);
  int s = (int)(t / nheads);
  long base = ((long)s * nheads + h) * HD + 2 * p;
  float vr = (float)buf[base];
  float vi = (float)buf[base + 1];
  float c  = fc[(long)s * (HD / 2) + p];
  float sn = fs[(long)s * (HD / 2) + p];
  buf[base]     = (__bf16)(vr * c - vi * sn);
  buf[base + 1] = (__bf16)(vr * sn + vi * c);
}

// V: bf16 [SEQ][NKVH][HD] -> Vt: bf16 [NKVH][HD][SEQ]
__global__ void vtrans_kernel(const __bf16* __restrict__ V, __bf16* __restrict__ Vt) {
  long i = (long)blockIdx.x * blockDim.x + threadIdx.x;
  long total = (long)NKVH * HD * SEQ;
  if (i >= total) return;
  int s = (int)(i % SEQ);
  long t = i / SEQ;
  int d = (int)(t % HD);
  int kvh = (int)(t / HD);
  Vt[i] = V[(long)s * KVDIM + (long)kvh * HD + d];
}

// ---------------- WMMA GEMM:  C[M][N] = A[M][K] * BT[N][K]^T ----------------
// 128 threads = 4 waves; wave tile 64(M) x 64(N); block tile 128 x 128.
// ~32 FLOP per cache byte at K-step 32.

__global__ __launch_bounds__(128) void gemm_bf16_kernel(
    const __bf16* __restrict__ A, const __bf16* __restrict__ BT,
    __bf16* __restrict__ Cb, float* __restrict__ Cf, int M, int N, int Kd)
{
  const int wave = threadIdx.x >> 5;
  const int lane = threadIdx.x & 31;
  const int half = lane >> 4;
  const int l16  = lane & 15;
  const int wr = wave & 1;
  const int wc = wave >> 1;
  const int m0 = blockIdx.y * 128 + wr * 64;
  const int n0 = blockIdx.x * 128 + wc * 64;

  v8f zero = {};
  v8f acc[4][4];
#pragma unroll
  for (int i = 0; i < 4; ++i)
#pragma unroll
    for (int j = 0; j < 4; ++j) acc[i][j] = zero;

  const __bf16* arow[4];
  const __bf16* brow[4];
#pragma unroll
  for (int i = 0; i < 4; ++i) arow[i] = A + (long)(m0 + i * 16 + l16) * Kd;
#pragma unroll
  for (int j = 0; j < 4; ++j) brow[j] = BT + (long)(n0 + j * 16 + l16) * Kd;

  for (int kb = 0; kb < Kd; kb += 32) {
    const int aoff = kb + half * 16;
    v16bf a[4];
#pragma unroll
    for (int i = 0; i < 4; ++i)
      a[i] = cat16(*(const v8bf*)(arow[i] + aoff), *(const v8bf*)(arow[i] + aoff + 8));
    __builtin_prefetch(arow[0] + kb + 256, 0, 3);   // global_prefetch_b8, keep near
    __builtin_prefetch(brow[0] + kb + 256, 0, 3);
#pragma unroll
    for (int j = 0; j < 4; ++j) {
      const __bf16* bp = brow[j] + kb + half * 16;
      v16bf b = cat16(*(const v8bf*)bp, *(const v8bf*)(bp + 8));
#pragma unroll
      for (int i = 0; i < 4; ++i) acc[i][j] = wmma_bf16(a[i], b, acc[i][j]);
    }
  }

  // C/D layout: vgpr r -> row = (lane/16)*8 + r ; col = lane%16
  if (Cb) {
#pragma unroll
    for (int i = 0; i < 4; ++i)
#pragma unroll
      for (int j = 0; j < 4; ++j) {
        const long col = n0 + j * 16 + l16;
#pragma unroll
        for (int r = 0; r < 8; ++r) {
          const long row = m0 + i * 16 + half * 8 + r;
          Cb[row * N + col] = (__bf16)acc[i][j][r];
        }
      }
  } else {
#pragma unroll
    for (int i = 0; i < 4; ++i)
#pragma unroll
      for (int j = 0; j < 4; ++j) {
        const long col = n0 + j * 16 + l16;
#pragma unroll
        for (int r = 0; r < 8; ++r) {
          const long row = m0 + i * 16 + half * 8 + r;
          Cf[row * N + col] = acc[i][j][r];
        }
      }
  }
}

// ---------------- Flash attention (sliding window, GQA) ----------------
// One wave per (q-head, 16-query tile). 4 waves per block.
// K tiles staged in LDS via ASYNCcnt-tracked async copies (double-buffered).
// Q[SEQ][NQH*HD], K[SEQ][NKVH*HD], Vt[NKVH][HD][SEQ], O[SEQ][NQH*HD]

__global__ __launch_bounds__(128) void attn_kernel(
    const __bf16* __restrict__ Q, const __bf16* __restrict__ Kmat,
    const __bf16* __restrict__ Vt, __bf16* __restrict__ O)
{
#ifdef HAVE_ASYNC_LDS
  __shared__ __bf16 ldsK[4][2][32][136];   // per-wave double-buffered K tile, padded rows
#endif
  __shared__ __bf16 ldsP[4][16][32];       // per-wave P-tile bounce buffer
  const int wave = threadIdx.x >> 5;
  const int lane = threadIdx.x & 31;
  const int half = lane >> 4;
  const int l16  = lane & 15;
  const int h    = blockIdx.y;     // query head
  const int kvh  = h >> 2;         // GQA: kv head = h / 4
  const int qbase = (blockIdx.x * 4 + wave) * 16;

  // Q A-fragments (16 x 128), resident for the whole kv loop
  v16bf qa[4];
  const __bf16* qrow = Q + (long)(qbase + l16) * DIM + (long)h * HD;
#pragma unroll
  for (int s = 0; s < 4; ++s) {
    const int off = s * 32 + half * 16;
    qa[s] = cat16(*(const v8bf*)(qrow + off), *(const v8bf*)(qrow + off + 8));
  }

  v8f zero = {};
  v8f oacc[8];
#pragma unroll
  for (int j = 0; j < 8; ++j) oacc[j] = zero;
  float mrow[8], lrow[8];
#pragma unroll
  for (int r = 0; r < 8; ++r) { mrow[r] = -1e30f; lrow[r] = 0.f; }

  const float scale = 0.08838834764831845f;  // 1/sqrt(128)
  const int qmax = qbase + 15;
  const int kb0 = (qbase > WINDOW) ? ((qbase - WINDOW) & ~31) : 0;

#ifdef HAVE_ASYNC_LDS
  // Copy K rows [kb, kb+32) x 128 bf16 into ldsK[wave][buf]; 16 async b128 ops.
  auto issue_ktile = [&](int kb, int buf) {
    const int rsel = lane >> 4;            // lane -> one of 2 rows per op
    const int cb   = (lane & 15) * 16;     // byte offset within 256B row payload
#pragma unroll
    for (int i = 0; i < 16; ++i) {
      const int row = i * 2 + rsel;
      const char* gp =
          (const char*)(Kmat + (long)(kb + row) * KVDIM + (long)kvh * HD) + cb;
      char* lp = (char*)&ldsK[wave][buf][row][0] + cb;
      __builtin_amdgcn_global_load_async_to_lds_b128(
          (as1_v4i*)gp, (as3_v4i*)lp, 0, 0);
    }
  };
  issue_ktile(kb0, (kb0 >> 5) & 1);
#endif

  for (int kb = kb0; kb <= qmax; kb += 32) {
#ifdef HAVE_ASYNC_LDS
    const int cur = (kb >> 5) & 1;
    if (kb + 32 <= qmax) {
      issue_ktile(kb + 32, cur ^ 1);            // prefetch next tile
      __builtin_amdgcn_s_wait_asynccnt(16);     // current tile landed
    } else {
      __builtin_amdgcn_s_wait_asynccnt(0);
    }
    const __bf16* krow0 = &ldsK[wave][cur][l16][0];
    const __bf16* krow1 = &ldsK[wave][cur][l16 + 16][0];
#else
    const __bf16* krow0 = Kmat + (long)(kb + l16) * KVDIM + (long)kvh * HD;
    const __bf16* krow1 = krow0 + (long)16 * KVDIM;
#endif
    // ---- scores S = Q K^T for 32 keys (two 16-key column groups) ----
    v8f sacc0 = zero, sacc1 = zero;
#pragma unroll
    for (int s = 0; s < 4; ++s) {
      const int off = s * 32 + half * 16;
      v16bf b0 = cat16(*(const v8bf*)(krow0 + off), *(const v8bf*)(krow0 + off + 8));
      v16bf b1 = cat16(*(const v8bf*)(krow1 + off), *(const v8bf*)(krow1 + off + 8));
      sacc0 = wmma_bf16(qa[s], b0, sacc0);
      sacc1 = wmma_bf16(qa[s], b1, sacc1);
    }
    // ---- mask + online softmax (rows live in 16-lane halves) ----
    float alpha[8], p0[8], p1[8];
#pragma unroll
    for (int r = 0; r < 8; ++r) {
      const int row = qbase + half * 8 + r;
      const int c0 = kb + l16;
      const int c1 = kb + 16 + l16;
      float s0 = sacc0[r] * scale;
      float s1 = sacc1[r] * scale;
      if (c0 > row || c0 < row - WINDOW) s0 = -1e30f;
      if (c1 > row || c1 < row - WINDOW) s1 = -1e30f;
      float mx = fmaxf(s0, s1);
      mx = fmaxf(mx, __shfl_xor(mx, 1, 32));
      mx = fmaxf(mx, __shfl_xor(mx, 2, 32));
      mx = fmaxf(mx, __shfl_xor(mx, 4, 32));
      mx = fmaxf(mx, __shfl_xor(mx, 8, 32));
      const float mnew = fmaxf(mrow[r], mx);
      alpha[r] = __expf(mrow[r] - mnew);
      const float e0 = __expf(s0 - mnew);
      const float e1 = __expf(s1 - mnew);
      float rs = e0 + e1;
      rs += __shfl_xor(rs, 1, 32);
      rs += __shfl_xor(rs, 2, 32);
      rs += __shfl_xor(rs, 4, 32);
      rs += __shfl_xor(rs, 8, 32);
      lrow[r] = lrow[r] * alpha[r] + rs;
      mrow[r] = mnew;
      p0[r] = e0;
      p1[r] = e1;
    }
#pragma unroll
    for (int j = 0; j < 8; ++j)
#pragma unroll
      for (int r = 0; r < 8; ++r) oacc[j][r] *= alpha[r];

    // ---- transpose P (D-layout -> A-layout) through per-wave LDS ----
#pragma unroll
    for (int r = 0; r < 8; ++r) {
      ldsP[wave][half * 8 + r][l16]      = (__bf16)p0[r];
      ldsP[wave][half * 8 + r][16 + l16] = (__bf16)p1[r];
    }
    v16bf pf = cat16(*(const v8bf*)(&ldsP[wave][l16][half * 16]),
                     *(const v8bf*)(&ldsP[wave][l16][half * 16 + 8]));

    // ---- O += P * V  (Vt rows give contiguous B-fragments) ----
    const __bf16* vbase = Vt + ((long)kvh * HD + l16) * SEQ + kb + half * 16;
#pragma unroll
    for (int j = 0; j < 8; ++j) {
      const __bf16* vp = vbase + (long)(j * 16) * SEQ;
      v16bf vf = cat16(*(const v8bf*)vp, *(const v8bf*)(vp + 8));
      oacc[j] = wmma_bf16(pf, vf, oacc[j]);
    }
  }

  float inv[8];
#pragma unroll
  for (int r = 0; r < 8; ++r) inv[r] = 1.f / lrow[r];
#pragma unroll
  for (int j = 0; j < 8; ++j) {
    const long col = (long)h * HD + j * 16 + l16;
#pragma unroll
    for (int r = 0; r < 8; ++r) {
      const long row = qbase + half * 8 + r;
      O[row * DIM + col] = (__bf16)(oacc[j][r] * inv[r]);
    }
  }
}

// ---------------- host orchestration ----------------

extern "C" void kernel_launch(void* const* d_in, const int* in_sizes, int n_in,
                              void* d_out, int out_size, void* d_ws, size_t ws_size,
                              hipStream_t stream) {
  const float* x  = (const float*)d_in[0];
  const float* fc = (const float*)d_in[1];
  const float* fs = (const float*)d_in[2];
  const float* wq = (const float*)d_in[3];
  const float* wk = (const float*)d_in[4];
  const float* wv = (const float*)d_in[5];
  const float* wo = (const float*)d_in[6];
  float* out = (float*)d_out;

  char* ws = (char*)d_ws;
  const size_t MB = 1ull << 20;
  __bf16* Xb   = (__bf16*)(ws + 0 * MB);     // [2048][4096]      16 MB
  __bf16* WqT  = (__bf16*)(ws + 16 * MB);    // [4096][4096]      32 MB
  __bf16* WkT  = (__bf16*)(ws + 48 * MB);    // [1024][4096]       8 MB
  __bf16* WvT  = (__bf16*)(ws + 56 * MB);    // [1024][4096]       8 MB
  __bf16* WoT  = (__bf16*)(ws + 64 * MB);    // [4096][4096]      32 MB
  __bf16* Qb   = (__bf16*)(ws + 96 * MB);    // [2048][4096]      16 MB
  __bf16* Kbuf = (__bf16*)(ws + 112 * MB);   // [2048][1024]       4 MB
  __bf16* Vb   = (__bf16*)(ws + 116 * MB);   // [2048][1024]       4 MB
  __bf16* Vtb  = (__bf16*)(ws + 120 * MB);   // [8][128][2048]     4 MB
  __bf16* Ob   = (__bf16*)(ws + 124 * MB);   // [2048][4096]      16 MB

  auto g1 = [](long n) { return (unsigned)((n + 255) / 256); };

  // fp32 -> bf16 conversions (+ weight transposes to [N][K])
  cvt_kernel<<<g1((long)SEQ * DIM), 256, 0, stream>>>(x, Xb, (long)SEQ * DIM);
  cvt_t_kernel<<<g1((long)DIM * DIM),   256, 0, stream>>>(wq, WqT, DIM, DIM);
  cvt_t_kernel<<<g1((long)DIM * KVDIM), 256, 0, stream>>>(wk, WkT, DIM, KVDIM);
  cvt_t_kernel<<<g1((long)DIM * KVDIM), 256, 0, stream>>>(wv, WvT, DIM, KVDIM);
  cvt_t_kernel<<<g1((long)DIM * DIM),   256, 0, stream>>>(wo, WoT, DIM, DIM);

  dim3 blk(128);
  // QKV projections (block tile 128x128)
  gemm_bf16_kernel<<<dim3(DIM / 128, SEQ / 128), blk, 0, stream>>>(
      Xb, WqT, Qb, nullptr, SEQ, DIM, DIM);
  gemm_bf16_kernel<<<dim3(KVDIM / 128, SEQ / 128), blk, 0, stream>>>(
      Xb, WkT, Kbuf, nullptr, SEQ, KVDIM, DIM);
  gemm_bf16_kernel<<<dim3(KVDIM / 128, SEQ / 128), blk, 0, stream>>>(
      Xb, WvT, Vb, nullptr, SEQ, KVDIM, DIM);

  // RoPE on Q and K, transpose V per kv-head
  rope_kernel<<<g1((long)SEQ * NQH * (HD / 2)),  256, 0, stream>>>(Qb, fc, fs, NQH);
  rope_kernel<<<g1((long)SEQ * NKVH * (HD / 2)), 256, 0, stream>>>(Kbuf, fc, fs, NKVH);
  vtrans_kernel<<<g1((long)NKVH * HD * SEQ), 256, 0, stream>>>(Vb, Vtb);

  // Sliding-window flash attention
  attn_kernel<<<dim3(SEQ / 64, NQH), blk, 0, stream>>>(Qb, Kbuf, Vtb, Ob);

  // Output projection -> fp32 result
  gemm_bf16_kernel<<<dim3(DIM / 128, SEQ / 128), blk, 0, stream>>>(
      Ob, WoT, nullptr, out, SEQ, DIM, DIM);

  (void)in_sizes; (void)n_in; (void)out_size; (void)ws_size;
}
